// Decoder_38225208934505
// MI455X (gfx1250) — matrix operand order repaired and checked by reading, not compiled, not run on hardware
//
#include <hip/hip_runtime.h>
#include <hip/hip_bf16.h>
#include <cstdint>
#include <cstddef>

// ---------------------------------------------------------------------------
// Autoregressive 2-layer GRU decoder for MI455X (gfx1250, wave32, WMMA).
//
// Design (latency-bound, 256 sequential steps):
//  * persistent kernel: 128 workgroups x 128 threads (4 waves), each wave owns
//    a 16-row batch tile and iterates all 256 steps with hidden state in LDS.
//  * layer-0 input GEMM folded into a 33x384 token table (exact; x has only
//    33 possible values), so only W_hh0, W_ih1, W_hh1 (288KB f16, packed into
//    WMMA B-fragment layout) live in LDS; +4 waves * (h0+h1 f16 tiles 8KB)
//    = 320KB LDS exactly (CDNA5 per-WG max).
//  * all matmuls via v_wmma_f32_16x16x32_f16 (f32 accumulate).
//  * argmax + act-softmax computed in WMMA D-layout with shfl_xor reductions.
//  * duration softmax (over time axis) done by a small second-pass kernel.
// ---------------------------------------------------------------------------

#define HIDDEN 128
#define GATES  384
#define TSTEPS 256
#define OUTC   33
#define BATCH  8192

typedef __attribute__((ext_vector_type(16))) _Float16 v16h;
typedef __attribute__((ext_vector_type(8)))  float    v8f;

struct B32x8 { uint4 lo, hi; };

static __device__ __forceinline__ v16h make_frag(const void* p0, const void* p1) {
  B32x8 u;
  u.lo = *(const uint4*)p0;
  u.hi = *(const uint4*)p1;
  return __builtin_bit_cast(v16h, u);
}

static __device__ __forceinline__ v8f wmma16(v16h a, v16h b, v8f c) {
  // D = A(16x32 f16) * B(32x16 f16) + C(16x16 f32)
  return __builtin_amdgcn_wmma_f32_16x16x32_f16(false, a, false, b, (short)0, c,
                                                false, false);
}

static __device__ __forceinline__ float fsigmoid(float x) {
  return 1.0f / (1.0f + __expf(-x));
}
static __device__ __forceinline__ float ftanh(float x) {
  return 2.0f * fsigmoid(2.0f * x) - 1.0f;
}

// A-fragment (16x32, f16) from a row-major 16x128 f16 tile in LDS.
// ISA 16-bit A layout: lane<16 holds row M=lane, dwords0-3 = K 0..7,
// dwords4-7 = K 16..23; lanes>=16 get K+8 (all contiguous per group).
static __device__ __forceinline__ v16h load_afrag(const _Float16* tile, int kc,
                                                  int lane) {
  const int m  = lane & 15;
  const int hh = lane >> 4;
  const _Float16* p0 = tile + m * HIDDEN + kc * 32 + hh * 8;
  return make_frag(p0, p0 + 16);
}

// B-fragment (32x16, f16) from a packed lane-major buffer (1KB / fragment):
// lane l owns 32 bytes at frag_base + 32*l (two b128 reads).
static __device__ __forceinline__ v16h load_bfrag(const char* base, int frag,
                                                  int lane) {
  const char* p = base + (frag << 10) + lane * 32;
  return make_frag(p, p + 16);
}

// ---------------------------------------------------------------------------
// Prep kernel 1: token table  g0[t][g] = b_ih0[g] + x_t . w_ih0[g,:]
// where x_t = [emb[t], 1.0] for t<32, and t==32 is the SOS start [emb[0], 0.0].
// ---------------------------------------------------------------------------
__global__ void build_g0_kernel(const float* __restrict__ emb,
                                const float* __restrict__ wih0,
                                const float* __restrict__ bih0,
                                float* __restrict__ g0) {
  int idx = blockIdx.x * blockDim.x + threadIdx.x;
  if (idx >= 33 * GATES) return;
  int tkn  = idx / GATES;
  int g    = idx - tkn * GATES;
  int te   = (tkn == 32) ? 0 : tkn;
  float fl = (tkn == 32) ? 0.0f : 1.0f;
  const float* wr = wih0 + g * HIDDEN;
  float acc = bih0[g];
  for (int k = 0; k < HIDDEN - 1; ++k) acc += emb[te * (HIDDEN - 1) + k] * wr[k];
  acc += fl * wr[HIDDEN - 1];
  g0[idx] = acc;
}

// ---------------------------------------------------------------------------
// Prep kernel 2: pack W (N x 128 row-major f32, representing B = W^T) into
// f16 WMMA B-fragments. Fragment (nt,kc): lane l<16 -> N=nt*16+l,
// K = kc*32 + 2(v&3) + 16*(v>>2) (+1 in hi half of dword); lanes>=16 -> K += 8.
// ---------------------------------------------------------------------------
__global__ void pack_b_kernel(const float* __restrict__ w,
                              unsigned* __restrict__ dst, int ntiles,
                              int nvalid) {
  int idx = blockIdx.x * blockDim.x + threadIdx.x;
  if (idx >= ntiles * 4 * 256) return;
  int f   = idx >> 8;          // fragment index
  int rem = idx & 255;
  int l   = rem >> 3;          // lane
  int v   = rem & 7;           // dword within lane
  int nt  = f >> 2, kc = f & 3;
  int n   = nt * 16 + (l & 15);
  int k   = kc * 32 + ((l >> 4) << 3) + ((v >> 2) << 4) + 2 * (v & 3);
  float a = (n < nvalid) ? w[n * HIDDEN + k] : 0.0f;
  float b = (n < nvalid) ? w[n * HIDDEN + k + 1] : 0.0f;
  unsigned short ua = __builtin_bit_cast(unsigned short, (_Float16)a);
  unsigned short ub = __builtin_bit_cast(unsigned short, (_Float16)b);
  dst[idx] = ((unsigned)ub << 16) | (unsigned)ua;
}

// ---------------------------------------------------------------------------
// Main persistent decode kernel.
// ---------------------------------------------------------------------------
__global__ __launch_bounds__(128) void gru_decode_kernel(
    const float* __restrict__ g0tab,     // [33][384]
    const float* __restrict__ packAll,   // packed Whh0|Wih1|Whh1 (contiguous)
    const float* __restrict__ packFc,    // packed fc fragments (3*4 frags)
    const float* __restrict__ hiddenInit,// [2][8192][128] f32
    const float* __restrict__ bhh0, const float* __restrict__ bih1,
    const float* __restrict__ bhh1, const float* __restrict__ fcb,
    float* __restrict__ outLogp, float* __restrict__ outProb) {
  extern __shared__ char smem[];
  char* wWhh0 = smem;            // 96KB (24 ntiles * 4 kc * 1KB)
  char* wWih1 = smem + 98304;    // 96KB
  char* wWhh1 = smem + 196608;   // 96KB

  const int tid  = threadIdx.x;
  const int lane = tid & 31;
  const int wave = tid >> 5;
  _Float16* h0tile = (_Float16*)(smem + 294912 + wave * 8192);
  _Float16* h1tile = (_Float16*)((char*)h0tile + 4096);

  // ---- stage all three weight matrices into LDS (18432 uint4) ----
  {
    const uint4* src = (const uint4*)packAll;
    uint4* dst = (uint4*)smem;
    for (int i = tid; i < 18432; i += 128) dst[i] = src[i];
  }
  // ---- per-wave hidden-state tiles (f16, row-major 16x128) ----
  const int b0 = blockIdx.x * 64 + wave * 16;
  {
    const float* s0 = hiddenInit + (size_t)b0 * HIDDEN;
    const float* s1 = hiddenInit + (size_t)BATCH * HIDDEN + (size_t)b0 * HIDDEN;
    for (int i = lane; i < 16 * HIDDEN; i += 32) {
      h0tile[i] = (_Float16)s0[i];
      h1tile[i] = (_Float16)s1[i];
    }
  }
  __syncthreads();

  const int nl = lane & 15;  // column within D-tile
  const int hh = lane >> 4;  // row-half select (rows 0-7 vs 8-15)
  int tok[8];
#pragma unroll
  for (int r = 0; r < 8; ++r) tok[r] = 32;  // SOS pseudo-token

  for (int t = 0; t < TSTEPS; ++t) {
    // A-fragments of h0(old), h1(old)
    v16h ah0[4], ah1[4];
#pragma unroll
    for (int kc = 0; kc < 4; ++kc) {
      ah0[kc] = load_afrag(h0tile, kc, lane);
      ah1[kc] = load_afrag(h1tile, kc, lane);
    }

    // ---------------- layer 0: gh = h0 @ Whh0^T + b_hh0, gi from table ----
    for (int c = 0; c < 8; ++c) {
      float b_r = bhh0[c * 16 + nl];
      float b_z = bhh0[128 + c * 16 + nl];
      float b_n = bhh0[256 + c * 16 + nl];
      v8f hr, hz, hn;
#pragma unroll
      for (int i = 0; i < 8; ++i) { hr[i] = b_r; hz[i] = b_z; hn[i] = b_n; }
#pragma unroll
      for (int kc = 0; kc < 4; ++kc) {
        hr = wmma16(ah0[kc], load_bfrag(wWhh0, (c)      * 4 + kc, lane), hr);
        hz = wmma16(ah0[kc], load_bfrag(wWhh0, (8 + c)  * 4 + kc, lane), hz);
        hn = wmma16(ah0[kc], load_bfrag(wWhh0, (16 + c) * 4 + kc, lane), hn);
      }
#pragma unroll
      for (int r = 0; r < 8; ++r) {
        int m = r + hh * 8;
        const float* grow = g0tab + tok[r] * GATES + c * 16 + nl;
        float ir = grow[0], iz = grow[128], in = grow[256];
        float rg = fsigmoid(ir + hr[r]);
        float zg = fsigmoid(iz + hz[r]);
        float ng = ftanh(in + rg * hn[r]);
        _Float16* hp = h0tile + m * HIDDEN + c * 16 + nl;
        float hold = (float)(*hp);
        *hp = (_Float16)((1.0f - zg) * ng + zg * hold);
      }
    }

    v16h ah0n[4];
#pragma unroll
    for (int kc = 0; kc < 4; ++kc) ah0n[kc] = load_afrag(h0tile, kc, lane);

    // ---------------- layer 1: full GRU cell on h0new / h1old ----
    for (int c = 0; c < 8; ++c) {
      float bir = bih1[c * 16 + nl], biz = bih1[128 + c * 16 + nl],
            bin = bih1[256 + c * 16 + nl];
      float bhr = bhh1[c * 16 + nl], bhz = bhh1[128 + c * 16 + nl],
            bhn = bhh1[256 + c * 16 + nl];
      v8f gir, giz, gin, ghr, ghz, ghn;
#pragma unroll
      for (int i = 0; i < 8; ++i) {
        gir[i] = bir; giz[i] = biz; gin[i] = bin;
        ghr[i] = bhr; ghz[i] = bhz; ghn[i] = bhn;
      }
#pragma unroll
      for (int kc = 0; kc < 4; ++kc) {
        gir = wmma16(ah0n[kc], load_bfrag(wWih1, (c)      * 4 + kc, lane), gir);
        giz = wmma16(ah0n[kc], load_bfrag(wWih1, (8 + c)  * 4 + kc, lane), giz);
        gin = wmma16(ah0n[kc], load_bfrag(wWih1, (16 + c) * 4 + kc, lane), gin);
        ghr = wmma16(ah1[kc],  load_bfrag(wWhh1, (c)      * 4 + kc, lane), ghr);
        ghz = wmma16(ah1[kc],  load_bfrag(wWhh1, (8 + c)  * 4 + kc, lane), ghz);
        ghn = wmma16(ah1[kc],  load_bfrag(wWhh1, (16 + c) * 4 + kc, lane), ghn);
      }
#pragma unroll
      for (int r = 0; r < 8; ++r) {
        int m = r + hh * 8;
        float rg = fsigmoid(gir[r] + ghr[r]);
        float zg = fsigmoid(giz[r] + ghz[r]);
        float ng = ftanh(gin[r] + rg * ghn[r]);
        _Float16* hp = h1tile + m * HIDDEN + c * 16 + nl;
        float hold = (float)(*hp);
        *hp = (_Float16)((1.0f - zg) * ng + zg * hold);
      }
    }

    // ---------------- fc head: pred = h1new @ fc_w^T + fc_b (cols 0..32) ----
    v16h ah1n[4];
#pragma unroll
    for (int kc = 0; kc < 4; ++kc) ah1n[kc] = load_afrag(h1tile, kc, lane);
    v8f p[3];
#pragma unroll
    for (int nt = 0; nt < 3; ++nt) {
      int n = nt * 16 + nl;
      float fb = (n < OUTC) ? fcb[n] : 0.0f;
#pragma unroll
      for (int i = 0; i < 8; ++i) p[nt][i] = fb;
#pragma unroll
      for (int kc = 0; kc < 4; ++kc)
        p[nt] = wmma16(ah1n[kc],
                       load_bfrag((const char*)packFc, nt * 4 + kc, lane), p[nt]);
    }

    // ---------------- argmax over cols 0..31 (per row, in D-layout) ----
#pragma unroll
    for (int r = 0; r < 8; ++r) {
      float v = p[0][r];
      int ix = nl;
      if (p[1][r] > v) { v = p[1][r]; ix = 16 + nl; }
#pragma unroll
      for (int mk = 1; mk <= 8; mk <<= 1) {
        float vo = __shfl_xor(v, mk, 32);
        int io = __shfl_xor(ix, mk, 32);
        if (vo > v || (vo == v && io < ix)) { v = vo; ix = io; }
      }
      tok[r] = ix;  // lane holds tokens for exactly its own 8 rows
    }

    // ---------------- act softmax / log_softmax + output stores ----
#pragma unroll
    for (int r = 0; r < 8; ++r) {
      float mx = fmaxf(p[0][r], p[1][r]);
#pragma unroll
      for (int mk = 1; mk <= 8; mk <<= 1) mx = fmaxf(mx, __shfl_xor(mx, mk, 32));
      float e0 = __expf(p[0][r] - mx);
      float e1 = __expf(p[1][r] - mx);
      float s = e0 + e1;
#pragma unroll
      for (int mk = 1; mk <= 8; mk <<= 1) s += __shfl_xor(s, mk, 32);
      float inv = 1.0f / s;
      float lgs = __logf(s);
      int m = r + hh * 8;
      size_t rowbase = ((size_t)(b0 + m) * TSTEPS + (size_t)t) * OUTC;
      outLogp[rowbase + nl]      = p[0][r] - mx - lgs;
      outLogp[rowbase + 16 + nl] = p[1][r] - mx - lgs;
      outProb[rowbase + nl]      = e0 * inv;
      outProb[rowbase + 16 + nl] = e1 * inv;
      if (nl == 0) outLogp[rowbase + 32] = p[2][r];  // raw duration logit
    }
  }
}

// ---------------------------------------------------------------------------
// Pass 2: softmax of duration logits over the TIME axis, per batch row.
// ---------------------------------------------------------------------------
__global__ void durations_kernel(float* __restrict__ outLogp,
                                 float* __restrict__ outProb) {
  int b = blockIdx.x * blockDim.x + threadIdx.x;
  if (b >= BATCH) return;
  size_t base = (size_t)b * TSTEPS * OUTC + 32;
  float mx = -3.4e38f;
  for (int t = 0; t < TSTEPS; ++t)
    mx = fmaxf(mx, outLogp[base + (size_t)t * OUTC]);
  float s = 0.0f;
  for (int t = 0; t < TSTEPS; ++t)
    s += __expf(outLogp[base + (size_t)t * OUTC] - mx);
  float inv = 1.0f / s;
  for (int t = 0; t < TSTEPS; ++t) {
    float d = __expf(outLogp[base + (size_t)t * OUTC] - mx) * inv;
    outLogp[base + (size_t)t * OUTC] = d;
    outProb[base + (size_t)t * OUTC] = d;
  }
}

// ---------------------------------------------------------------------------
extern "C" void kernel_launch(void* const* d_in, const int* in_sizes, int n_in,
                              void* d_out, int out_size, void* d_ws,
                              size_t ws_size, hipStream_t stream) {
  (void)in_sizes; (void)n_in; (void)out_size; (void)ws_size;
  const float* hidden = (const float*)d_in[1];
  const float* emb    = (const float*)d_in[2];
  const float* wih0   = (const float*)d_in[3];
  const float* whh0   = (const float*)d_in[4];
  const float* bih0   = (const float*)d_in[5];
  const float* bhh0   = (const float*)d_in[6];
  const float* wih1   = (const float*)d_in[7];
  const float* whh1   = (const float*)d_in[8];
  const float* bih1   = (const float*)d_in[9];
  const float* bhh1   = (const float*)d_in[10];
  const float* fcw    = (const float*)d_in[11];
  const float* fcb    = (const float*)d_in[12];

  float* ws    = (float*)d_ws;
  float* g0    = ws;             // 33*384           = 12672 f32
  float* pWhh0 = ws + 12672;     // 24 ntiles packed = 24576 dw
  float* pWih1 = ws + 37248;
  float* pWhh1 = ws + 61824;
  float* pFc   = ws + 86400;     // 3 ntiles packed  =  3072 dw

  float* outLogp = (float*)d_out;
  float* outProb = outLogp + (size_t)BATCH * TSTEPS * OUTC;

  build_g0_kernel<<<(33 * GATES + 255) / 256, 256, 0, stream>>>(emb, wih0, bih0,
                                                                g0);
  pack_b_kernel<<<(24 * 4 * 256) / 256, 256, 0, stream>>>(whh0,
                                                          (unsigned*)pWhh0, 24,
                                                          GATES);
  pack_b_kernel<<<(24 * 4 * 256) / 256, 256, 0, stream>>>(wih1,
                                                          (unsigned*)pWih1, 24,
                                                          GATES);
  pack_b_kernel<<<(24 * 4 * 256) / 256, 256, 0, stream>>>(whh1,
                                                          (unsigned*)pWhh1, 24,
                                                          GATES);
  pack_b_kernel<<<(3 * 4 * 256) / 256, 256, 0, stream>>>(fcw, (unsigned*)pFc, 3,
                                                         OUTC);

  const int SMEM = 294912 + 4 * 8192;  // 320KB: 3 weight mats + 4 wave tiles
  (void)hipFuncSetAttribute((const void*)gru_decode_kernel,
                            hipFuncAttributeMaxDynamicSharedMemorySize, SMEM);
  gru_decode_kernel<<<BATCH / 64, 128, SMEM, stream>>>(
      g0, pWhh0, pFc, hidden, bhh0, bih1, bhh1, fcb, outLogp, outProb);

  durations_kernel<<<(BATCH + 255) / 256, 256, 0, stream>>>(outLogp, outProb);
}
